// TransformerBlock_14181982012036
// MI455X (gfx1250) — compile-verified, matrix operand order checked by reading, not run on hardware
//
#include <hip/hip_runtime.h>

// Problem dims (fixed by the reference)
#define B_   8
#define S_   1024
#define D_   768
#define H_   12
#define DH_  64
#define FF_  3072
#define M_   (B_ * S_)   // 8192 rows

typedef __bf16 bf16_t;
typedef __attribute__((ext_vector_type(4)))  __bf16 bf16x4;
typedef __attribute__((ext_vector_type(8)))  __bf16 bf16x8;
typedef __attribute__((ext_vector_type(16))) __bf16 bf16x16;
typedef __attribute__((ext_vector_type(4)))  float  f32x4;
typedef __attribute__((ext_vector_type(8)))  float  f32x8;

static __device__ __forceinline__ bf16_t f2bf(float f) {
  unsigned u = __builtin_bit_cast(unsigned, f);
  u += 0x7fffu + ((u >> 16) & 1u);             // round-to-nearest-even
  unsigned short s = (unsigned short)(u >> 16);
  return __builtin_bit_cast(bf16_t, s);
}

static __device__ __forceinline__ bf16x16 cat8(bf16x8 lo, bf16x8 hi) {
  return __builtin_shufflevector(lo, hi, 0,1,2,3,4,5,6,7,8,9,10,11,12,13,14,15);
}

#define WMMA_BF16(A_, B_v, C_) \
  __builtin_amdgcn_wmma_f32_16x16x32_bf16(false, (A_), false, (B_v), (short)0, (C_), false, false)

// ---------------------------------------------------------------------------
// Tiled GEMM: Out[M,N] = A[M,K] * W[K,N] + bias  (bf16 WMMA, f32 accumulate)
// Block: 256 threads (8 waves), 128x128 tile, K-step 32.
// Wave w -> 32x64 sub-tile: 2 (M) x 4 (N) accumulators of 16x16.
// ABF: A operand is bf16 in memory (else f32, converted while staging).
// OBF: write bf16 output (else f32).  RELU: fuse max(0, .).
// ---------------------------------------------------------------------------
template<int ABF, int OBF, int RELU>
__global__ __launch_bounds__(256)
void gemm_kernel(const void* __restrict__ Ain, const float* __restrict__ W,
                 const float* __restrict__ bias, void* __restrict__ Out,
                 int M, int N, int K)
{
  (void)M;
  const int LDA = 40;   // 32 + 8 bf16 pad (80B row stride, 16B aligned)
  const int LDB = 40;
  __shared__ bf16_t sA[128 * 40];   // [row][k]
  __shared__ bf16_t sB[128 * 40];   // transposed: [n][k]

  const int tid  = threadIdx.x;
  const int lane = tid & 31;
  const int wid  = tid >> 5;
  const int half = lane >> 4;       // lanes 0-15 vs 16-31
  const int l16  = lane & 15;
  const int wm   = (wid >> 1) * 32; // 4 waves along M
  const int wn   = (wid & 1) * 64;  // 2 waves along N
  const int m0   = blockIdx.y * 128;
  const int n0   = blockIdx.x * 128;

  f32x8 acc[2][4];
  for (int i = 0; i < 2; i++)
    for (int j = 0; j < 4; j++) acc[i][j] = (f32x8)0.0f;

  for (int k0 = 0; k0 < K; k0 += 32) {
    // ---- stage A tile (convert f32->bf16 if needed) ----
    if (ABF) {
      const bf16_t* A = (const bf16_t*)Ain;
      for (int i = 0; i < 2; i++) {
        int c = tid + i * 256;                 // 512 chunks of 8 bf16
        int row = c >> 2, kc = (c & 3) * 8;
        bf16x8 v = *(const bf16x8*)(A + (size_t)(m0 + row) * K + k0 + kc);
        *(bf16x8*)&sA[row * LDA + kc] = v;
      }
    } else {
      const float* A = (const float*)Ain;
      for (int i = 0; i < 4; i++) {
        int c = tid + i * 256;                 // 1024 chunks of 4 f32
        int row = c >> 3, kc = (c & 7) * 4;
        f32x4 v = *(const f32x4*)(A + (size_t)(m0 + row) * K + k0 + kc);
        bf16x4 b4;
        b4[0] = f2bf(v.x); b4[1] = f2bf(v.y); b4[2] = f2bf(v.z); b4[3] = f2bf(v.w);
        *(bf16x4*)&sA[row * LDA + kc] = b4;
      }
    }
    // ---- stage B tile transposed: sB[n][k] = W[k0+k][n0+n] ----
    for (int i = 0; i < 4; i++) {
      int c = tid + i * 256;                   // 1024 chunks of 4 f32
      int kr = c >> 5, nc = (c & 31) * 4;
      f32x4 v = *(const f32x4*)(W + (size_t)(k0 + kr) * N + n0 + nc);
      sB[(nc + 0) * LDB + kr] = f2bf(v.x);
      sB[(nc + 1) * LDB + kr] = f2bf(v.y);
      sB[(nc + 2) * LDB + kr] = f2bf(v.z);
      sB[(nc + 3) * LDB + kr] = f2bf(v.w);
    }
    __syncthreads();

    // ---- fragments + 8 WMMAs ----
    bf16x16 af[2];
    for (int mi = 0; mi < 2; mi++) {
      int r = wm + mi * 16 + l16;
      bf16x8 lo = *(const bf16x8*)&sA[r * LDA + half * 8];
      bf16x8 hi = *(const bf16x8*)&sA[r * LDA + 16 + half * 8];
      af[mi] = cat8(lo, hi);
    }
    for (int nf = 0; nf < 4; nf++) {
      int ccol = wn + nf * 16 + l16;
      bf16x8 lo = *(const bf16x8*)&sB[ccol * LDB + half * 8];
      bf16x8 hi = *(const bf16x8*)&sB[ccol * LDB + 16 + half * 8];
      bf16x16 bfr = cat8(lo, hi);
      acc[0][nf] = WMMA_BF16(af[0], bfr, acc[0][nf]);
      acc[1][nf] = WMMA_BF16(af[1], bfr, acc[1][nf]);
    }
    __syncthreads();
  }

  // ---- epilogue: bias (+relu), f32 or bf16 store ----
  for (int mi = 0; mi < 2; mi++) {
    for (int nf = 0; nf < 4; nf++) {
      int col = n0 + wn + nf * 16 + l16;
      float bv = bias ? bias[col] : 0.0f;
      for (int r = 0; r < 8; r++) {
        int row = m0 + wm + mi * 16 + half * 8 + r;
        float v = acc[mi][nf][r] + bv;
        if (RELU) v = fmaxf(v, 0.0f);
        if (OBF) ((bf16_t*)Out)[(size_t)row * N + col] = f2bf(v);
        else     ((float*)Out)[(size_t)row * N + col] = v;
      }
    }
  }
}

// ---------------------------------------------------------------------------
// Flash attention: grid (S/64, B*H), 128 threads (4 waves).
// Wave w owns 16 query rows. KV tiles of 64 staged in LDS (V transposed).
// Online softmax; P re-laid-out through per-wave LDS scratch into A-fragments.
// Output written f32 (input to out-projection GEMM), already un-head-split.
// ---------------------------------------------------------------------------
__global__ __launch_bounds__(128)
void attn_kernel(const bf16_t* __restrict__ Q, const bf16_t* __restrict__ Kb,
                 const bf16_t* __restrict__ Vb, const float* __restrict__ mask,
                 float* __restrict__ Out)
{
  const int LDK = 72;  // 64 + 8 bf16 pad (144B stride)
  __shared__ bf16_t sK [64 * 72];     // [kv][d]
  __shared__ bf16_t sVt[64 * 72];     // [d][kv]
  __shared__ bf16_t sP [4][16 * 72];  // per-wave P scratch [qrow][kv]

  const int tid  = threadIdx.x;
  const int lane = tid & 31;
  const int w    = tid >> 5;
  const int half = lane >> 4;
  const int l16  = lane & 15;
  const int q0   = blockIdx.x * 64;
  const int bh   = blockIdx.y;
  const int b    = bh / H_, h = bh % H_;
  const size_t baseRow = (size_t)b * S_;
  const int hcol = h * DH_;

  // Q fragments (16 rows x 64 d), kept in registers for the whole kv loop
  bf16x16 qa[2];
  {
    size_t gr = baseRow + q0 + w * 16 + l16;
    const bf16_t* qp = Q + gr * D_ + hcol;
    for (int ks = 0; ks < 2; ks++) {
      bf16x8 lo = *(const bf16x8*)(qp + ks * 32 + half * 8);
      bf16x8 hi = *(const bf16x8*)(qp + ks * 32 + 16 + half * 8);
      qa[ks] = cat8(lo, hi);
    }
  }

  float m_i[8], l_i[8];
  f32x8 o[4];
  for (int r = 0; r < 8; r++) { m_i[r] = -1e30f; l_i[r] = 0.0f; }
  for (int nf = 0; nf < 4; nf++) o[nf] = (f32x8)0.0f;

  for (int kv0 = 0; kv0 < S_; kv0 += 64) {
    // stage K tile row-major, V tile transposed
    for (int i = 0; i < 4; i++) {
      int c = tid + i * 128;                 // 512 chunks of 8 bf16
      int row = c >> 3, ch = (c & 7) * 8;
      bf16x8 kvv = *(const bf16x8*)(Kb + (baseRow + kv0 + row) * D_ + hcol + ch);
      *(bf16x8*)&sK[row * LDK + ch] = kvv;
      bf16x8 vv = *(const bf16x8*)(Vb + (baseRow + kv0 + row) * D_ + hcol + ch);
      for (int j = 0; j < 8; j++) sVt[(ch + j) * LDK + row] = vv[j];
    }
    __syncthreads();

    // S = (Q K^T) * 1/sqrt(64) + mask * -1e9   (B-frag reads sK rows directly)
    f32x8 s[4];
    for (int nf = 0; nf < 4; nf++) {
      f32x8 a = (f32x8)0.0f;
      int kcol = nf * 16 + l16;
      bf16x8 lo0 = *(const bf16x8*)&sK[kcol * LDK + half * 8];
      bf16x8 hi0 = *(const bf16x8*)&sK[kcol * LDK + 16 + half * 8];
      a = WMMA_BF16(qa[0], cat8(lo0, hi0), a);
      bf16x8 lo1 = *(const bf16x8*)&sK[kcol * LDK + 32 + half * 8];
      bf16x8 hi1 = *(const bf16x8*)&sK[kcol * LDK + 48 + half * 8];
      a = WMMA_BF16(qa[1], cat8(lo1, hi1), a);
      float mv = mask[(size_t)b * S_ + kv0 + nf * 16 + l16] * -1e9f;
      for (int r = 0; r < 8; r++) a[r] = a[r] * 0.125f + mv;
      s[nf] = a;
    }

    // online softmax (row stats live replicated across each 16-lane group)
    for (int r = 0; r < 8; r++) {
      float v = fmaxf(fmaxf(s[0][r], s[1][r]), fmaxf(s[2][r], s[3][r]));
      v = fmaxf(v, __shfl_xor(v, 1, 32));
      v = fmaxf(v, __shfl_xor(v, 2, 32));
      v = fmaxf(v, __shfl_xor(v, 4, 32));
      v = fmaxf(v, __shfl_xor(v, 8, 32));
      float mn = fmaxf(m_i[r], v);
      float alpha = __expf(m_i[r] - mn);
      m_i[r] = mn;
      float rs = 0.0f;
      for (int nf = 0; nf < 4; nf++) {
        float p = __expf(s[nf][r] - mn);
        s[nf][r] = p;
        rs += p;
      }
      rs += __shfl_xor(rs, 1, 32);
      rs += __shfl_xor(rs, 2, 32);
      rs += __shfl_xor(rs, 4, 32);
      rs += __shfl_xor(rs, 8, 32);
      l_i[r] = l_i[r] * alpha + rs;
      for (int nf = 0; nf < 4; nf++) o[nf][r] *= alpha;
    }

    // C-layout -> A-fragment relayout of P through per-wave LDS scratch
    for (int nf = 0; nf < 4; nf++)
      for (int r = 0; r < 8; r++)
        sP[w][(half * 8 + r) * LDK + nf * 16 + l16] = f2bf(s[nf][r]);
    __builtin_amdgcn_fence(__ATOMIC_ACQ_REL, "wavefront");

    bf16x16 pa[2];
    for (int ks = 0; ks < 2; ks++) {
      bf16x8 lo = *(const bf16x8*)&sP[w][l16 * LDK + ks * 32 + half * 8];
      bf16x8 hi = *(const bf16x8*)&sP[w][l16 * LDK + ks * 32 + 16 + half * 8];
      pa[ks] = cat8(lo, hi);
    }
    // O += P V
    for (int nf = 0; nf < 4; nf++) {
      int dcol = nf * 16 + l16;
      bf16x8 lo0 = *(const bf16x8*)&sVt[dcol * LDK + half * 8];
      bf16x8 hi0 = *(const bf16x8*)&sVt[dcol * LDK + 16 + half * 8];
      o[nf] = WMMA_BF16(pa[0], cat8(lo0, hi0), o[nf]);
      bf16x8 lo1 = *(const bf16x8*)&sVt[dcol * LDK + 32 + half * 8];
      bf16x8 hi1 = *(const bf16x8*)&sVt[dcol * LDK + 48 + half * 8];
      o[nf] = WMMA_BF16(pa[1], cat8(lo1, hi1), o[nf]);
    }
    __syncthreads();
  }

  // finalize: divide by l, write f32 [B*S, D] (heads re-interleaved)
  for (int nf = 0; nf < 4; nf++) {
    for (int r = 0; r < 8; r++) {
      size_t row = baseRow + q0 + w * 16 + half * 8 + r;
      Out[row * D_ + hcol + nf * 16 + l16] = o[nf][r] / l_i[r];
    }
  }
}

// ---------------------------------------------------------------------------
// Fused residual + LayerNorm over rows of length 768. One block per row.
// out = LN(a + b) * g + be
// ---------------------------------------------------------------------------
__global__ __launch_bounds__(256)
void ln_kernel(const float* __restrict__ a, const float* __restrict__ b,
               const float* __restrict__ g, const float* __restrict__ be,
               float* __restrict__ out)
{
  __shared__ float red[256];
  const int row = blockIdx.x;
  const int tid = threadIdx.x;
  const float* ap = a + (size_t)row * D_;
  const float* bp = b + (size_t)row * D_;

  float loc[3];
  float sum = 0.0f;
  for (int i = 0; i < 3; i++) {
    float v = ap[tid + i * 256] + bp[tid + i * 256];
    loc[i] = v;
    sum += v;
  }
  red[tid] = sum; __syncthreads();
  for (int st = 128; st > 0; st >>= 1) {
    if (tid < st) red[tid] += red[tid + st];
    __syncthreads();
  }
  float mean = red[0] * (1.0f / D_);
  __syncthreads();

  float vs = 0.0f;
  for (int i = 0; i < 3; i++) { float d = loc[i] - mean; vs += d * d; }
  red[tid] = vs; __syncthreads();
  for (int st = 128; st > 0; st >>= 1) {
    if (tid < st) red[tid] += red[tid + st];
    __syncthreads();
  }
  float rstd = rsqrtf(red[0] * (1.0f / D_) + 1e-6f);

  for (int i = 0; i < 3; i++) {
    int col = tid + i * 256;
    out[(size_t)row * D_ + col] = (loc[i] - mean) * rstd * g[col] + be[col];
  }
}

// ---------------------------------------------------------------------------
extern "C" void kernel_launch(void* const* d_in, const int* in_sizes, int n_in,
                              void* d_out, int out_size, void* d_ws, size_t ws_size,
                              hipStream_t stream)
{
  (void)in_sizes; (void)n_in; (void)out_size; (void)ws_size;
  const float* x    = (const float*)d_in[0];
  const float* mask = (const float*)d_in[1];
  const float* Wq = (const float*)d_in[2];  const float* bq  = (const float*)d_in[3];
  const float* Wk = (const float*)d_in[4];  const float* bk  = (const float*)d_in[5];
  const float* Wv = (const float*)d_in[6];  const float* bv  = (const float*)d_in[7];
  const float* Wo = (const float*)d_in[8];  const float* bo  = (const float*)d_in[9];
  const float* W1 = (const float*)d_in[10]; const float* b1  = (const float*)d_in[11];
  const float* W2 = (const float*)d_in[12]; const float* b2  = (const float*)d_in[13];
  const float* g1 = (const float*)d_in[14]; const float* be1 = (const float*)d_in[15];
  const float* g2 = (const float*)d_in[16]; const float* be2 = (const float*)d_in[17];
  float* out = (float*)d_out;

  char* ws = (char*)d_ws;
  const size_t szBF = (size_t)M_ * D_ * sizeof(bf16_t);   // 12.58 MB
  const size_t szF  = (size_t)M_ * D_ * sizeof(float);    // 25.17 MB
  bf16_t* Qbf  = (bf16_t*)(ws);
  bf16_t* Kbf  = (bf16_t*)(ws + szBF);
  bf16_t* Vbf  = (bf16_t*)(ws + 2 * szBF);
  float*  attn = (float*)(ws + 3 * szBF);
  float*  y    = (float*)(ws + 3 * szBF + szF);           // reused for FFN2 out
  float*  out1 = (float*)(ws + 3 * szBF + 2 * szF);
  bf16_t* hbf  = (bf16_t*)(ws + 3 * szBF + 3 * szF);      // M*FF bf16 = 50.3 MB

  dim3 blk(256);
  dim3 gD (D_  / 128, M_ / 128);   // 6 x 64
  dim3 gFF(FF_ / 128, M_ / 128);   // 24 x 64

  // QKV projections (bf16 outputs feed attention)
  gemm_kernel<0, 1, 0><<<gD, blk, 0, stream>>>(x, Wq, bq, Qbf, M_, D_, D_);
  gemm_kernel<0, 1, 0><<<gD, blk, 0, stream>>>(x, Wk, bk, Kbf, M_, D_, D_);
  gemm_kernel<0, 1, 0><<<gD, blk, 0, stream>>>(x, Wv, bv, Vbf, M_, D_, D_);

  // Flash attention -> f32 attn [B*S, D]
  attn_kernel<<<dim3(S_ / 64, B_ * H_), dim3(128), 0, stream>>>(Qbf, Kbf, Vbf, mask, attn);

  // Output projection, residual + LN1
  gemm_kernel<0, 0, 0><<<gD, blk, 0, stream>>>(attn, Wo, bo, y, M_, D_, D_);
  ln_kernel<<<dim3(M_), blk, 0, stream>>>(x, y, g1, be1, out1);

  // FFN: relu(out1 W1 + b1) W2 + b2, residual + LN2
  gemm_kernel<0, 1, 1><<<gFF, blk, 0, stream>>>(out1, W1, b1, hbf, M_, FF_, D_);
  gemm_kernel<1, 0, 0><<<gD,  blk, 0, stream>>>(hbf,  W2, b2, y,   M_, D_, FF_);
  ln_kernel<<<dim3(M_), blk, 0, stream>>>(out1, y, g2, be2, out);
}
